// GroupQueryAttention_68848325755148
// MI455X (gfx1250) — matrix operand order repaired and checked by reading, not compiled, not run on hardware
//
#include <hip/hip_runtime.h>
#include <hip/hip_bf16.h>

// ---------------- problem constants ----------------
constexpr int kB  = 2;
constexpr int kS  = 2048;
constexpr int kE  = 2048;
constexpr int kH  = 32;
constexpr int kG  = 8;
constexpr int kGH = 4;     // kH / kG
constexpr int kD  = 64;    // kE / kH
constexpr int kKV = 512;   // kG * kD
constexpr int kM  = kB * kS;   // 4096 rows in all GEMMs

typedef __attribute__((ext_vector_type(16))) __bf16 v16bf;
typedef __attribute__((ext_vector_type(8)))  float  v8f;
typedef __attribute__((ext_vector_type(4)))  unsigned int u32x4;
typedef __attribute__((ext_vector_type(8)))  int i32x8;
typedef __attribute__((ext_vector_type(4)))  int i32x4;

__device__ __forceinline__ unsigned short f32_to_bf16(float f) {
  unsigned int u = __builtin_bit_cast(unsigned int, f);
  unsigned int r = u + 0x7FFFu + ((u >> 16) & 1u);   // round-to-nearest-even
  return (unsigned short)(r >> 16);
}

// ---- Tensor Data Mover: async 2D bf16 tile load (64x64) global -> LDS ----
// D# per CDNA5 ISA ch.8: group0 = {count, lds_addr, global_addr, type=2},
// group1 = {data_size=2B, tensor_dim0/1, tile 64x64, tensor_dim0_stride}.
// This toolchain exposes the 6-arg builtin: (g0, g1, g2, g3, g_extra, cpol).
__device__ __forceinline__ void tdm_load_tile64(unsigned int lds_off,
                                                const unsigned short* gptr,
                                                unsigned int stride_elems,
                                                unsigned int dim1) {
  unsigned long long ga = (unsigned long long)gptr;
  u32x4 g0;
  g0[0] = 1u;                                            // count=1, no gather
  g0[1] = lds_off;                                       // LDS byte address
  g0[2] = (unsigned int)(ga & 0xFFFFFFFFu);              // global_addr[31:0]
  g0[3] = (unsigned int)((ga >> 32) & 0x1FFFFFFu)        // global_addr[56:32]
          | (2u << 30);                                  // type=2 ("image")
  i32x8 g1;
  g1[0] = (int)(1u << 16);                               // wg_mask=0, data_size=2B
  g1[1] = (int)((stride_elems & 0xFFFFu) << 16);         // tensor_dim0[15:0]
  g1[2] = (int)((stride_elems >> 16) | ((dim1 & 0xFFFFu) << 16)); // dim0[31:16]|dim1[15:0]
  g1[3] = (int)((dim1 >> 16) | (64u << 16));             // dim1[31:16] | tile_dim0=64
  g1[4] = (int)64u;                                      // tile_dim1=64, tile_dim2=0
  g1[5] = (int)stride_elems;                             // tensor_dim0_stride[31:0]
  g1[6] = 0;                                             // stride0 hi | stride1 lo
  g1[7] = 0;                                             // stride1 hi
  i32x4 z4 = {0, 0, 0, 0};
  i32x8 z8 = {0, 0, 0, 0, 0, 0, 0, 0};
  __builtin_amdgcn_tensor_load_to_lds(g0, g1, z4, z4, z8, 0);
}

// ---------------- f32 -> bf16 convert ----------------
__global__ __launch_bounds__(256) void cvt_f32_bf16(const float* __restrict__ in,
                                                    unsigned short* __restrict__ out,
                                                    int n) {
  int i = blockIdx.x * 256 + threadIdx.x;
  if (i < n) out[i] = f32_to_bf16(in[i]);
}

// ---------------- tiled bf16 GEMM: C[M,N] = A[M,K] @ W[N,K]^T + bias ----------------
// block: 256 threads = 8 waves; tile BM=128, BN=128, BK=32.
// V_TRANS: scatter-store output as Vt[b][g][d][s] for the attention V operand.
template <bool OUT_F32, bool V_TRANS>
__global__ __launch_bounds__(256) void gemm_bf16(const unsigned short* __restrict__ A,
                                                 const unsigned short* __restrict__ W,
                                                 const float* __restrict__ bias,
                                                 void* __restrict__ Cout,
                                                 int M, int N, int K) {
  constexpr int BM = 128, BN = 128, BK = 32;
  __shared__ __align__(16) unsigned short As[BM][BK];
  __shared__ __align__(16) unsigned short Ws[BN][BK];

  const int tid  = threadIdx.x;
  const int lane = tid & 31;
  const int wave = tid >> 5;
  const int bm = blockIdx.y * BM;
  const int bn = blockIdx.x * BN;
  const int wm = (wave & 3) * 32;
  const int wn = (wave >> 2) * 64;

  v8f acc[2][4] = {};

  const int mrow = lane & 15;
  const int akb  = (lane < 16) ? 0 : 8;    // A K-pair base
  const int bkb  = (lane < 16) ? 0 : 8;    // B K-pair base

  for (int k0 = 0; k0 < K; k0 += BK) {
    __syncthreads();
    #pragma unroll
    for (int i = 0; i < 2; ++i) {
      int idx = tid + i * 256;
      int r = idx >> 2;
      int c = (idx & 3) * 8;
      *(uint4*)&As[r][c] = *(const uint4*)&A[(size_t)(bm + r) * K + k0 + c];
      *(uint4*)&Ws[r][c] = *(const uint4*)&W[(size_t)(bn + r) * K + k0 + c];
      if (k0 + BK < K) {
        __builtin_prefetch(&A[(size_t)(bm + r) * K + k0 + BK + c], 0, 0);
        __builtin_prefetch(&W[(size_t)(bn + r) * K + k0 + BK + c], 0, 0);
      }
    }
    __syncthreads();

    union Frag { v16bf v; unsigned int w[8]; };
    Frag af[2], bf[4];
    #pragma unroll
    for (int m = 0; m < 2; ++m) {
      const unsigned int* arow = (const unsigned int*)&As[wm + m * 16 + mrow][0];
      #pragma unroll
      for (int j = 0; j < 8; ++j) {
        int k = (j < 4) ? (akb + 2 * j) : (16 + akb + 2 * (j - 4));
        af[m].w[j] = arow[k >> 1];
      }
    }
    #pragma unroll
    for (int n = 0; n < 4; ++n) {
      const unsigned int* wrow = (const unsigned int*)&Ws[wn + n * 16 + mrow][0];
      #pragma unroll
      for (int j = 0; j < 8; ++j)
        bf[n].w[j] = wrow[((j < 4) ? (bkb + 2 * j) : (16 + bkb + 2 * (j - 4))) >> 1];
    }
    #pragma unroll
    for (int m = 0; m < 2; ++m)
      #pragma unroll
      for (int n = 0; n < 4; ++n)
        acc[m][n] = __builtin_amdgcn_wmma_f32_16x16x32_bf16(
            false, af[m].v, false, bf[n].v, (short)0, acc[m][n], false, false);
  }

  // epilogue: lane holds N = lane&15, VGPR j holds M = j + (lane<16?0:8)
  const int nlane = lane & 15;
  const int mbase = (lane < 16) ? 0 : 8;
  #pragma unroll
  for (int m = 0; m < 2; ++m)
    #pragma unroll
    for (int n = 0; n < 4; ++n) {
      int col = bn + wn + n * 16 + nlane;
      float bsv = bias[col];
      #pragma unroll
      for (int j = 0; j < 8; ++j) {
        int row = bm + wm + m * 16 + mbase + j;
        float val = acc[m][n][j] + bsv;
        if (OUT_F32) {
          ((float*)Cout)[(size_t)row * N + col] = val;
        } else if (V_TRANS) {
          // row = b*kS + s ; col = g*kD + d  ->  Vt[((b*kG+g)*kD + d)*kS + s]
          int bb = row >> 11, s = row & (kS - 1);
          int gg = col >> 6,  d = col & (kD - 1);
          ((unsigned short*)Cout)[(((size_t)(bb * kG + gg) * kD + d) * kS) + s] =
              f32_to_bf16(val);
        } else {
          ((unsigned short*)Cout)[(size_t)row * N + col] = f32_to_bf16(val);
        }
      }
    }
}

// ---------------- grouped-query flash attention ----------------
// grid: B*H*(S/64) blocks, 256 threads (8 waves).
// K and Vt tiles streamed into LDS by the Tensor Data Mover (TENSORcnt).
__global__ __launch_bounds__(256) void gqa_attn(const unsigned short* __restrict__ Q,
                                                const unsigned short* __restrict__ Km,
                                                const unsigned short* __restrict__ Vt,
                                                unsigned short* __restrict__ O) {
  constexpr int nQB = kS / 64;                 // 32
  const int qb = blockIdx.x % nQB;
  const int h  = (blockIdx.x / nQB) % kH;
  const int b  = blockIdx.x / (nQB * kH);
  const int g  = h / kGH;
  const int qbase = qb * 64;
  const float scale = 0.125f;                  // 1/sqrt(64)

  __shared__ __align__(16) unsigned short Qs[64][64];
  __shared__ __align__(16) unsigned short Ks[64][64];   // [t][d]
  __shared__ __align__(16) unsigned short Vts[64][64];  // [d][t]
  __shared__ __align__(16) unsigned short Ps[64][64];   // [q][t]
  __shared__ float Ss[64][64];
  __shared__ float mrun[64], lrun[64], arow[64];

  const int tid  = threadIdx.x;
  const int lane = tid & 31;
  const int wave = tid >> 5;
  const int r0 = (wave & 3) * 16;              // query rows of this wave
  const int c0 = (wave >> 2) * 32;             // key cols (scores) / d cols (output)

  // load Q tile [64 x 64] once
  #pragma unroll
  for (int i = 0; i < 2; ++i) {
    int idx = tid + i * 256;
    int r = idx >> 3;
    int c = (idx & 7) * 8;
    *(uint4*)&Qs[r][c] =
        *(const uint4*)&Q[(size_t)(b * kS + qbase + r) * kE + h * kD + c];
  }
  if (tid < 64) { mrun[tid] = -1e30f; lrun[tid] = 0.0f; }

  const unsigned int ldsK = (unsigned int)(unsigned long long)&Ks[0][0];
  const unsigned int ldsV = (unsigned int)(unsigned long long)&Vts[0][0];

  v8f accO[2] = {};

  const int mrow = lane & 15;
  const int akb  = (lane < 16) ? 0 : 8;
  const int bkb  = (lane < 16) ? 0 : 8;
  const int nlane = lane & 15;
  const int mbase = (lane < 16) ? 0 : 8;

  union Frag { v16bf v; unsigned int w[8]; };

  for (int kt = 0; kt < kS / 64; ++kt) {
    __syncthreads();   // previous iteration done reading Ks/Vts/Ps
    if (wave == 0) {
      // K tile: rows t (stride kKV), cols d   -> Ks[t][d]
      tdm_load_tile64(ldsK, &Km[(size_t)(b * kS + kt * 64) * kKV + g * kD],
                      kKV, kS);
      // Vt tile: rows d (stride kS), cols t   -> Vts[d][t]
      tdm_load_tile64(ldsV, &Vt[((size_t)(b * kG + g) * kD) * kS + kt * 64],
                      kS, kD);
      __builtin_amdgcn_s_wait_tensorcnt(0);
    }
    __syncthreads();   // tiles visible to all waves

    // ---- scores: S = Q @ K^T (wave computes 16 rows x 32 keys) ----
    v8f sacc[2] = {};
    #pragma unroll
    for (int d0 = 0; d0 < 64; d0 += 32) {
      Frag af;
      const unsigned int* qrow = (const unsigned int*)&Qs[r0 + mrow][0];
      #pragma unroll
      for (int j = 0; j < 8; ++j) {
        int k = d0 + ((j < 4) ? (akb + 2 * j) : (16 + akb + 2 * (j - 4)));
        af.w[j] = qrow[k >> 1];
      }
      #pragma unroll
      for (int n = 0; n < 2; ++n) {
        Frag bf;
        const unsigned int* krow = (const unsigned int*)&Ks[c0 + n * 16 + nlane][0];
        #pragma unroll
        for (int j = 0; j < 8; ++j)
          bf.w[j] = krow[(d0 + ((j < 4) ? (bkb + 2 * j) : (16 + bkb + 2 * (j - 4)))) >> 1];
        sacc[n] = __builtin_amdgcn_wmma_f32_16x16x32_bf16(
            false, af.v, false, bf.v, (short)0, sacc[n], false, false);
      }
    }
    #pragma unroll
    for (int n = 0; n < 2; ++n)
      #pragma unroll
      for (int j = 0; j < 8; ++j)
        Ss[r0 + mbase + j][c0 + n * 16 + nlane] = sacc[n][j] * scale;
    __syncthreads();

    // ---- online softmax (one thread per query row) ----
    if (tid < 64) {
      float tmax = -1e30f;
      #pragma unroll 8
      for (int c = 0; c < 64; ++c) tmax = fmaxf(tmax, Ss[tid][c]);
      float mold = mrun[tid];
      float mnew = fmaxf(mold, tmax);
      float a = __expf(mold - mnew);
      float sum = 0.0f;
      #pragma unroll 8
      for (int c = 0; c < 64; ++c) {
        float e = __expf(Ss[tid][c] - mnew);
        Ps[tid][c] = f32_to_bf16(e);
        sum += e;
      }
      lrun[tid] = lrun[tid] * a + sum;
      mrun[tid] = mnew;
      arow[tid] = a;
    }
    __syncthreads();

    // ---- rescale running output, then O += P @ V ----
    float al[8];
    #pragma unroll
    for (int j = 0; j < 8; ++j) al[j] = arow[r0 + mbase + j];
    #pragma unroll
    for (int n = 0; n < 2; ++n)
      #pragma unroll
      for (int j = 0; j < 8; ++j)
        accO[n][j] *= al[j];

    #pragma unroll
    for (int t0 = 0; t0 < 64; t0 += 32) {
      Frag af;
      const unsigned int* prow = (const unsigned int*)&Ps[r0 + mrow][0];
      #pragma unroll
      for (int j = 0; j < 8; ++j) {
        int k = t0 + ((j < 4) ? (akb + 2 * j) : (16 + akb + 2 * (j - 4)));
        af.w[j] = prow[k >> 1];
      }
      #pragma unroll
      for (int n = 0; n < 2; ++n) {
        Frag bf;
        // B[k=t][n=d] = Vts[d][t]; pairs (t,t+1) contiguous -> single b32 loads
        const unsigned int* vrow =
            (const unsigned int*)&Vts[c0 + n * 16 + nlane][0];
        #pragma unroll
        for (int j = 0; j < 8; ++j)
          bf.w[j] = vrow[(t0 + ((j < 4) ? (bkb + 2 * j) : (16 + bkb + 2 * (j - 4)))) >> 1];
        accO[n] = __builtin_amdgcn_wmma_f32_16x16x32_bf16(
            false, af.v, false, bf.v, (short)0, accO[n], false, false);
      }
    }
  }

  // ---- finalize: divide by row sum, store bf16 ----
  #pragma unroll
  for (int n = 0; n < 2; ++n)
    #pragma unroll
    for (int j = 0; j < 8; ++j) {
      int r = r0 + mbase + j;
      float val = accO[n][j] / lrun[r];
      O[(size_t)(b * kS + qbase + r) * kE + h * kD + c0 + n * 16 + nlane] =
          f32_to_bf16(val);
    }
}

// ---------------- workspace layout (bytes, 256-aligned) ----------------
constexpr size_t SZ_XB  = (size_t)kM * kE * 2;    // 16 MB
constexpr size_t SZ_WQB = (size_t)kE * kE * 2;    // 8 MB
constexpr size_t SZ_WKB = (size_t)kKV * kE * 2;   // 2 MB
constexpr size_t SZ_QB  = (size_t)kM * kE * 2;
constexpr size_t SZ_KB  = (size_t)kM * kKV * 2;

constexpr size_t OFF_XB  = 0;
constexpr size_t OFF_WQB = OFF_XB + SZ_XB;
constexpr size_t OFF_WKB = OFF_WQB + SZ_WQB;
constexpr size_t OFF_WVB = OFF_WKB + SZ_WKB;
constexpr size_t OFF_WOB = OFF_WVB + SZ_WKB;
constexpr size_t OFF_QB  = OFF_WOB + SZ_WQB;
constexpr size_t OFF_KB  = OFF_QB + SZ_QB;
constexpr size_t OFF_VT  = OFF_KB + SZ_KB;
constexpr size_t OFF_OB  = OFF_VT + SZ_KB;

extern "C" void kernel_launch(void* const* d_in, const int* in_sizes, int n_in,
                              void* d_out, int out_size, void* d_ws, size_t ws_size,
                              hipStream_t stream) {
  const float* x  = (const float*)d_in[0];
  const float* Wq = (const float*)d_in[1];
  const float* bq = (const float*)d_in[2];
  const float* Wk = (const float*)d_in[3];
  const float* bk = (const float*)d_in[4];
  const float* Wv = (const float*)d_in[5];
  const float* bv = (const float*)d_in[6];
  const float* Wo = (const float*)d_in[7];
  const float* bo = (const float*)d_in[8];
  float* out = (float*)d_out;

  char* ws = (char*)d_ws;
  unsigned short* xb  = (unsigned short*)(ws + OFF_XB);
  unsigned short* Wqb = (unsigned short*)(ws + OFF_WQB);
  unsigned short* Wkb = (unsigned short*)(ws + OFF_WKB);
  unsigned short* Wvb = (unsigned short*)(ws + OFF_WVB);
  unsigned short* Wob = (unsigned short*)(ws + OFF_WOB);
  unsigned short* Qb  = (unsigned short*)(ws + OFF_QB);
  unsigned short* Kb  = (unsigned short*)(ws + OFF_KB);
  unsigned short* Vtb = (unsigned short*)(ws + OFF_VT);
  unsigned short* Ob  = (unsigned short*)(ws + OFF_OB);

  auto cvt = [&](const float* src, unsigned short* dst, int n) {
    cvt_f32_bf16<<<(n + 255) / 256, 256, 0, stream>>>(src, dst, n);
  };
  cvt(x,  xb,  kM * kE);
  cvt(Wq, Wqb, kE * kE);
  cvt(Wk, Wkb, kKV * kE);
  cvt(Wv, Wvb, kKV * kE);
  cvt(Wo, Wob, kE * kE);

  // projections: Q [4096,2048], K [4096,512], V -> transposed Vt[b][g][d][s]
  gemm_bf16<false, false><<<dim3(kE / 128, kM / 128), 256, 0, stream>>>(
      xb, Wqb, bq, Qb, kM, kE, kE);
  gemm_bf16<false, false><<<dim3(kKV / 128, kM / 128), 256, 0, stream>>>(
      xb, Wkb, bk, Kb, kM, kKV, kE);
  gemm_bf16<false, true><<<dim3(kKV / 128, kM / 128), 256, 0, stream>>>(
      xb, Wvb, bv, Vtb, kM, kKV, kE);

  // grouped-query flash attention (TDM-fed K/V tiles)
  gqa_attn<<<kB * kH * (kS / 64), 256, 0, stream>>>(Qb, Kb, Vtb, Ob);

  // output projection -> f32
  gemm_bf16<true, false><<<dim3(kE / 128, kM / 128), 256, 0, stream>>>(
      Ob, Wob, bo, out, kM, kE, kE);
}
